// MoE_8976481648982
// MI455X (gfx1250) — matrix operand order repaired and checked by reading, not compiled, not run on hardware
//
#include <hip/hip_runtime.h>

// CDNA5 / gfx1250: wave32, WMMA 16x16x32 f16 -> f32 accumulate.
typedef __attribute__((ext_vector_type(16))) _Float16 v16h;
typedef __attribute__((ext_vector_type(8)))  _Float16 v8h;
typedef __attribute__((ext_vector_type(8)))  float    v8f;

namespace {
constexpr int kB = 4096;   // batch
constexpr int kS = 32;     // stations
constexpr int kD = 256;    // feature dim (layer-1 K)
constexpr int kH = 512;    // hidden dim  (layer-1 N, layer-2 K and N)

constexpr int M_TILE  = 32;       // batch rows per workgroup (2 row-tiles)
constexpr int THREADS = 512;      // 16 wave32; each wave owns 2 N-tiles
constexpr int NT      = kH / 16;  // 32 column tiles (both layers have N = 512)
constexpr int XELEM   = M_TILE * kD / THREADS;  // 16 X elements per thread

// LDS strides in halves: multiple of 8 (16B-aligned v8h loads), rows spread banks
constexpr int XS_H = kD + 8;      // 264
constexpr int HS_H = kH + 8;      // 520
}

__device__ __forceinline__ float gelu_exact(float x) {
  return 0.5f * x * (1.0f + erff(x * 0.7071067811865475f));
}

__device__ __forceinline__ v8f zero_v8f() {
  v8f z;
  #pragma unroll
  for (int i = 0; i < 8; ++i) z[i] = 0.0f;
  return z;
}

// ---------------------------------------------------------------------------
// Weight packing: f32 [K][N] -> fragment-ordered split-f16.
// Fragment (kt, nt): 32 lanes x (16 hi halves | 16 lo halves) contiguous.
// Lane L covers column nt*16 + (L&15), K rows kt*32 + (L<16 ? 0 : 8) + (i&7) + 16*(i>>3).
// Hot-loop B-fragment load = 64B contiguous per lane (pure global_load_b128).
// ---------------------------------------------------------------------------
__global__ __launch_bounds__(256)
void pack_weights_kernel(const float* __restrict__ src, _Float16* __restrict__ dst,
                         int K, int N) {
  const int s = blockIdx.y;                     // matrix index (station)
  src += (size_t)s * K * N;
  dst += (size_t)s * K * N * 2;                 // hi+lo halves

  const int t    = blockIdx.x * 256 + threadIdx.x;  // one thread per (frag,lane)
  const int lane = t & 31;
  const int frag = t >> 5;
  const int nt   = frag & (NT - 1);
  const int kt   = frag >> 5;                   // NT == 32
  if (kt >= (K >> 5)) return;

  const int n  = nt * 16 + (lane & 15);
  const int kb = kt * 32 + ((lane >> 4) << 3);

  v16h hi, lo;
  #pragma unroll
  for (int i = 0; i < 16; ++i) {
    const float x = src[(size_t)(kb + (i & 7) + ((i >> 3) << 4)) * N + n];
    const _Float16 h = (_Float16)x;
    hi[i] = h;
    lo[i] = (_Float16)(x - (float)h);
  }
  _Float16* q = dst + ((size_t)frag * 32 + lane) * 32;
  *(v16h*)q        = hi;
  *(v16h*)(q + 16) = lo;
}

// A-fragment (16x32 f16) from an LDS half array: two contiguous 8-half runs.
__device__ __forceinline__ v16h load_a_frag(const _Float16* base, int stride,
                                            int row, int k0, int a_kb) {
  const _Float16* r = base + row * stride + k0 + a_kb;
  const v8h p0 = *(const v8h*)r;
  const v8h p1 = *(const v8h*)(r + 16);
  return __builtin_shufflevector(p0, p1, 0, 1, 2, 3, 4, 5, 6, 7,
                                         8, 9, 10, 11, 12, 13, 14, 15);
}

// B-fragment from packed weights: 64B contiguous per lane.
__device__ __forceinline__ void load_b_frag(const _Float16* __restrict__ packed,
                                            int kt, int nt, int lane,
                                            v16h& hi, v16h& lo) {
  const _Float16* p = packed + ((size_t)(kt * NT + nt) * 32 + lane) * 32;
  hi = *(const v16h*)p;
  lo = *(const v16h*)(p + 16);
}

// Markidis-style 3-term split product, f32 accumulate (~2^-22 residual).
__device__ __forceinline__ v8f wmma_split3(v16h ah, v16h al, v16h bh, v16h bl, v8f c) {
  c = __builtin_amdgcn_wmma_f32_16x16x32_f16(false, ah, false, bh, (short)0, c, false, false);
  c = __builtin_amdgcn_wmma_f32_16x16x32_f16(false, ah, false, bl, (short)0, c, false, false);
  c = __builtin_amdgcn_wmma_f32_16x16x32_f16(false, al, false, bh, (short)0, c, false, false);
  return c;
}

// out[b,k] = sum over 33 experts of gelu(gelu(x @ W1 + b1) @ W2 + b2)[b,k]
// (attention/gate stage is dead: softmax rows sum to 1, so the final einsum
//  'bij,bik->bk' reduces to the plain sum over experts).
__global__ __launch_bounds__(THREADS)
void moe_expert_sum_kernel(const float* __restrict__ infer_point,
                           const float* __restrict__ stations,
                           const float* __restrict__ b1s,
                           const float* __restrict__ b2s,
                           const float* __restrict__ ib1,
                           const float* __restrict__ ib2,
                           const _Float16* __restrict__ pW1s,
                           const _Float16* __restrict__ pW2s,
                           const _Float16* __restrict__ piW1,
                           const _Float16* __restrict__ piW2,
                           float* __restrict__ out)
{
  __shared__ __align__(16) _Float16 XsH[M_TILE * XS_H];
  __shared__ __align__(16) _Float16 XsL[M_TILE * XS_H];
  __shared__ __align__(16) _Float16 HsH[M_TILE * HS_H];
  __shared__ __align__(16) _Float16 HsL[M_TILE * HS_H];

  const int tid  = threadIdx.x;
  const int lane = tid & 31;
  const int wave = tid >> 5;                 // 16 waves, each owns 2 N-tiles
  const int b0   = blockIdx.x * M_TILE;

  const int  col    = lane & 15;             // N within tile; also A row select
  const bool hiHalf = (lane >= 16);
  const int  a_kb   = hiHalf ? 8 : 0;        // A fragment K sub-base (ISA 7.12.2)
  const int  c_r0   = hiHalf ? 8 : 0;        // C fragment row base (M = c_r0 + v)

  // This thread's X-tile slice coordinates (16 elements, strided by THREADS).
  int xm[XELEM], xd[XELEM];
  #pragma unroll
  for (int i = 0; i < XELEM; ++i) {
    const int idx = tid + i * THREADS;
    xm[i] = idx >> 8;                        // kD == 256
    xd[i] = idx & (kD - 1);
  }

  // Prefetch station 0's X tile into registers (overlaps with nothing yet,
  // but establishes the s+1 pipeline below).
  float xr[XELEM];
  #pragma unroll
  for (int i = 0; i < XELEM; ++i) {
    xr[i] = stations[((size_t)(b0 + xm[i]) * kS + 0) * kD + xd[i]];
  }

  v8f outAcc[2][2];                          // [row-tile][n-tile]
  #pragma unroll
  for (int rt = 0; rt < 2; ++rt)
    #pragma unroll
    for (int ct = 0; ct < 2; ++ct) outAcc[rt][ct] = zero_v8f();

  for (int s = 0; s <= kS; ++s) {            // 32 stations + 1 infer expert
    const bool st = (s < kS);
    const _Float16* pW1 = st ? (pW1s + (size_t)s * (2 * kD * kH)) : piW1;
    const _Float16* pW2 = st ? (pW2s + (size_t)s * (2 * kH * kH)) : piW2;
    const float*    B1  = st ? (b1s + (size_t)s * kH) : ib1;
    const float*    B2  = st ? (b2s + (size_t)s * kH) : ib2;

    __syncthreads();  // previous iteration fully done with Xs/Hs

    // Stage this station's X tile from registers (no global latency here).
    #pragma unroll
    for (int i = 0; i < XELEM; ++i) {
      const float x = xr[i];
      const _Float16 h = (_Float16)x;
      XsH[xm[i] * XS_H + xd[i]] = h;
      XsL[xm[i] * XS_H + xd[i]] = (_Float16)(x - (float)h);
    }
    __syncthreads();

    // Prefetch NEXT expert's X tile; completes under the two GEMMs below.
    if (s < kS) {
      #pragma unroll
      for (int i = 0; i < XELEM; ++i) {
        xr[i] = (s + 1 < kS)
            ? stations[((size_t)(b0 + xm[i]) * kS + (s + 1)) * kD + xd[i]]
            : infer_point[(size_t)(b0 + xm[i]) * kD + xd[i]];
      }
    }

    // ---------------- layer 1: h = gelu(X @ W1 + b1), K = 256 ---------------
    v8f acc[2][2];
    #pragma unroll
    for (int rt = 0; rt < 2; ++rt)
      #pragma unroll
      for (int ct = 0; ct < 2; ++ct) acc[rt][ct] = zero_v8f();

    for (int k0 = 0; k0 < kD; k0 += 32) {
      const int kt = k0 >> 5;
      v16h aH[2], aL[2];
      #pragma unroll
      for (int rt = 0; rt < 2; ++rt) {
        aH[rt] = load_a_frag(XsH, XS_H, rt * 16 + col, k0, a_kb);
        aL[rt] = load_a_frag(XsL, XS_H, rt * 16 + col, k0, a_kb);
      }
      #pragma unroll
      for (int ct = 0; ct < 2; ++ct) {
        v16h b_hi, b_lo;
        load_b_frag(pW1, kt, wave * 2 + ct, lane, b_hi, b_lo);
        #pragma unroll
        for (int rt = 0; rt < 2; ++rt)
          acc[rt][ct] = wmma_split3(aH[rt], aL[rt], b_hi, b_lo, acc[rt][ct]);
      }
    }

    #pragma unroll
    for (int ct = 0; ct < 2; ++ct) {
      const int n = wave * 32 + ct * 16 + col;
      const float bias = B1[n];
      #pragma unroll
      for (int rt = 0; rt < 2; ++rt) {
        #pragma unroll
        for (int v = 0; v < 8; ++v) {
          const float hval = gelu_exact(acc[rt][ct][v] + bias);
          const _Float16 hh = (_Float16)hval;
          const int row = rt * 16 + c_r0 + v;
          HsH[row * HS_H + n] = hh;
          HsL[row * HS_H + n] = (_Float16)(hval - (float)hh);
        }
      }
    }
    __syncthreads();

    // -------- layer 2: o = gelu(h @ W2 + b2), K = 512; accumulate sum -------
    #pragma unroll
    for (int rt = 0; rt < 2; ++rt)
      #pragma unroll
      for (int ct = 0; ct < 2; ++ct) acc[rt][ct] = zero_v8f();

    for (int k0 = 0; k0 < kH; k0 += 32) {
      const int kt = k0 >> 5;
      v16h aH[2], aL[2];
      #pragma unroll
      for (int rt = 0; rt < 2; ++rt) {
        aH[rt] = load_a_frag(HsH, HS_H, rt * 16 + col, k0, a_kb);
        aL[rt] = load_a_frag(HsL, HS_H, rt * 16 + col, k0, a_kb);
      }
      #pragma unroll
      for (int ct = 0; ct < 2; ++ct) {
        v16h b_hi, b_lo;
        load_b_frag(pW2, kt, wave * 2 + ct, lane, b_hi, b_lo);
        #pragma unroll
        for (int rt = 0; rt < 2; ++rt)
          acc[rt][ct] = wmma_split3(aH[rt], aL[rt], b_hi, b_lo, acc[rt][ct]);
      }
    }

    #pragma unroll
    for (int ct = 0; ct < 2; ++ct) {
      const float bias = B2[wave * 32 + ct * 16 + col];
      #pragma unroll
      for (int rt = 0; rt < 2; ++rt) {
        #pragma unroll
        for (int v = 0; v < 8; ++v) {
          outAcc[rt][ct][v] += gelu_exact(acc[rt][ct][v] + bias);
        }
      }
    }
  }

  // ------------------------------ store [B,H] ------------------------------
  #pragma unroll
  for (int ct = 0; ct < 2; ++ct) {
    const int n = wave * 32 + ct * 16 + col;
    #pragma unroll
    for (int rt = 0; rt < 2; ++rt) {
      #pragma unroll
      for (int v = 0; v < 8; ++v) {
        out[(size_t)(b0 + rt * 16 + c_r0 + v) * kH + n] = outAcc[rt][ct][v];
      }
    }
  }
}

extern "C" void kernel_launch(void* const* d_in, const int* in_sizes, int n_in,
                              void* d_out, int out_size, void* d_ws, size_t ws_size,
                              hipStream_t stream) {
  (void)in_sizes; (void)n_in; (void)out_size; (void)ws_size;
  const float* infer_point = (const float*)d_in[0];
  const float* stations    = (const float*)d_in[1];
  const float* W1s = (const float*)d_in[2];
  const float* b1s = (const float*)d_in[3];
  const float* W2s = (const float*)d_in[4];
  const float* b2s = (const float*)d_in[5];
  const float* iW1 = (const float*)d_in[6];
  const float* ib1 = (const float*)d_in[7];
  const float* iW2 = (const float*)d_in[8];
  const float* ib2 = (const float*)d_in[9];
  // d_in[10] (gW) / d_in[11] (gb): mathematically dead (softmax rows sum to 1).
  float* out = (float*)d_out;

  // Workspace layout (halves): packed split-f16 weights, ~52 MB total.
  _Float16* wsbase = (_Float16*)d_ws;
  _Float16* pW1s = wsbase;                                   // 32 * 2*256*512
  _Float16* pW2s = pW1s + (size_t)kS * 2 * kD * kH;          // 32 * 2*512*512
  _Float16* piW1 = pW2s + (size_t)kS * 2 * kH * kH;          //      2*256*512
  _Float16* piW2 = piW1 + (size_t)2 * kD * kH;               //      2*512*512

  dim3 blk(256);
  // One thread per (fragment, lane): K*N/16 threads per matrix.
  pack_weights_kernel<<<dim3((kD * kH / 16) / 256, kS), blk, 0, stream>>>(W1s, pW1s, kD, kH);
  pack_weights_kernel<<<dim3((kH * kH / 16) / 256, kS), blk, 0, stream>>>(W2s, pW2s, kH, kH);
  pack_weights_kernel<<<dim3((kD * kH / 16) / 256, 1),  blk, 0, stream>>>(iW1, piW1, kD, kH);
  pack_weights_kernel<<<dim3((kH * kH / 16) / 256, 1),  blk, 0, stream>>>(iW2, piW2, kH, kH);

  moe_expert_sum_kernel<<<dim3(kB / M_TILE), dim3(THREADS), 0, stream>>>(
      infer_point, stations, b1s, b2s, ib1, ib2,
      pW1s, pW2s, piW1, piW2, out);
}
